// BrushModel_16174846837057
// MI455X (gfx1250) — matrix operand order repaired and checked by reading, not compiled, not run on hardware
//
#include <hip/hip_runtime.h>

#define P_EXPERTS 16
#define H_DIM     128
#define M_TILE    256      // rows per block (2 bands of 16 per wave)
#define NTHREADS  256
#define S_BT      136      // f16 row stride for K=128 buffers (pad vs bank conflicts)
#define S_X       40       // f16 row stride for K=32 buffers  (pad vs bank conflicts)

typedef __attribute__((ext_vector_type(16))) _Float16 v16h;
typedef __attribute__((ext_vector_type(8)))  float    v8f;

union Frag16 {
    v16h v;
    unsigned int u[8];
    _Float16 h[16];
};

// One-shot f32 -> f16 weight conversion into workspace (L2-resident afterwards).
__global__ void cvt_w2_f16(const float* __restrict__ src, _Float16* __restrict__ dst, int n)
{
    int i = blockIdx.x * blockDim.x + threadIdx.x;
    if (i < n) dst[i] = (_Float16)src[i];
}

// Fully WMMA-ized fused 3-layer expert MLP (single barrier):
//   h1 = relu(x @ W1^T + b1)   WMMA, K padded DIN->32
//   h2 = relu(h1 @ W2^T + b2)  WMMA, K=128
//   o  = h2 @ W3^T + b3        WMMA, N padded DOUT->16
template<int DIN, int DOUT, bool PRE>
__global__ __launch_bounds__(NTHREADS)
void moe_mlp_fused(const float* __restrict__ x,      // [N, DIN]
                   const float* __restrict__ W1,     // [P, H, DIN]
                   const float* __restrict__ b1,     // [P, H]
                   const float* __restrict__ W2,     // [P, H, H] (f32 fallback)
                   const float* __restrict__ b2,     // [P, H]
                   const float* __restrict__ W3,     // [P, DOUT, H]
                   const float* __restrict__ b3,     // [P, DOUT]
                   const _Float16* __restrict__ W2h, // [P, H, H] f16 (PRE path)
                   float* __restrict__ out,          // [P*N, DOUT]
                   int N)
{
    __shared__ __align__(16) _Float16 sA  [M_TILE * S_BT]; // h1 then h2 (f16)
    __shared__ __align__(16) _Float16 sBt [H_DIM  * S_BT]; // W2[k][h] f16
    __shared__ __align__(16) _Float16 sXh [M_TILE * S_X];  // x, K padded to 32
    __shared__ __align__(16) _Float16 sW1t[H_DIM  * S_X];  // W1[n][k], K padded to 32
    __shared__ __align__(16) _Float16 sW3t[16     * S_BT]; // W3[o][k], rows padded to 16
    __shared__ float sB1[H_DIM];
    __shared__ float sB2[H_DIM];
    __shared__ float sB3[DOUT];

    const int tid  = threadIdx.x;
    const int lane = tid & 31;
    const int wave = tid >> 5;             // 0..7
    const int p    = blockIdx.y;           // expert id
    const int row0 = blockIdx.x * M_TILE;  // base row of this tile

    // ================= Phase 0: stage all operands ====================
    if (PRE) {
        // Async DMA: f16 W2 straight into LDS (ASYNCcnt), overlapped with
        // the staging VALU work below. 16B per lane per op.
        #pragma unroll
        for (int c = tid; c < (H_DIM * H_DIM) / 8; c += NTHREADS) {
            int k  = c >> 4;
            int h8 = (c & 15) << 3;
            unsigned int dst =
                (unsigned int)(unsigned long long)(const void*)&sBt[k * S_BT + h8];
            const _Float16* g = W2h + ((p << 14) + (k << 7) + h8);
            asm volatile("global_load_async_to_lds_b128 %0, %1, off"
                         :: "v"(dst), "v"(g) : "memory");
        }
    } else {
        for (int i = tid; i < H_DIM * H_DIM; i += NTHREADS) {
            int k = i >> 7, h = i & 127;
            sBt[k * S_BT + h] = (_Float16)W2[(p * H_DIM + k) * H_DIM + h];
        }
    }
    // x tile -> f16, K zero-padded to 32 (single writer per element)
    for (int i = tid; i < M_TILE * 32; i += NTHREADS) {
        int r = i >> 5, k = i & 31;
        sXh[r * S_X + k] = (k < DIN) ? (_Float16)x[(row0 + r) * DIN + k] : (_Float16)0.f;
    }
    // W1^T -> f16, K zero-padded to 32
    for (int i = tid; i < H_DIM * 32; i += NTHREADS) {
        int n = i >> 5, k = i & 31;
        sW1t[n * S_X + k] = (k < DIN) ? (_Float16)W1[p * H_DIM * DIN + n * DIN + k]
                                      : (_Float16)0.f;
    }
    // W3^T -> f16, output channels zero-padded to 16
    for (int i = tid; i < 16 * S_BT; i += NTHREADS) {
        int o = i / S_BT, k = i - o * S_BT;
        float v = (o < DOUT && k < H_DIM) ? W3[p * DOUT * H_DIM + o * H_DIM + k] : 0.f;
        sW3t[i] = (_Float16)v;
    }
    for (int i = tid; i < H_DIM; i += NTHREADS) {
        sB1[i] = b1[p * H_DIM + i];
        sB2[i] = b2[p * H_DIM + i];
    }
    if (tid < DOUT) sB3[tid] = b3[p * DOUT + tid];

    if (PRE) asm volatile("s_wait_asynccnt 0x0" ::: "memory");
    __syncthreads();   // the only barrier: everything below is wave-local

    // Per-lane fragment coordinates (CDNA5 16-bit WMMA layouts):
    //  A (16x32): lane<16 -> K {b..b+7, b+16..b+23}; lane>=16 -> +8
    //  B (32x16): col = lane&15; lane<16 -> K {b..b+7, b+16..b+23}; lane>=16 -> +8
    //  C/D      : lane<16 -> N=lane, VGPR r -> M=r ; lane>=16 -> N=lane-16, M=r+8
    const int mrow  = (wave << 4) + (lane & 15);           // band0 row; band1 = +128
    const int klo   = (lane < 16) ? 0 : 8;                 // A K base
    const int kbB   = (lane < 16) ? 0 : 8;                 // B K base
    const int bcol  = lane & 15;                           // B/D column
    const int crow0 = (wave << 4) + ((lane < 16) ? 0 : 8); // D row base

    // ================= Phase 1: layer 1 (WMMA, K=32) ==================
    // Issue all 16 independent WMMAs first, then consume: avoids the
    // WMMA->VALU hazard NOP bubbles of immediate consumption.
    {
        Frag16 xa0, xa1;
        const unsigned int* px0 = (const unsigned int*)&sXh[mrow * S_X + klo];
        const unsigned int* px1 = (const unsigned int*)&sXh[(mrow + 128) * S_X + klo];
        #pragma unroll
        for (int i = 0; i < 4; ++i) {
            xa0.u[i] = px0[i]; xa0.u[i + 4] = px0[i + 8];
            xa1.u[i] = px1[i]; xa1.u[i + 4] = px1[i + 8];
        }
        v8f h0acc[8];
        v8f h1acc[8];
        #pragma unroll
        for (int j = 0; j < 8; ++j) {
            Frag16 b;
            const unsigned int* pb =
                (const unsigned int*)&sW1t[(j * 16 + bcol) * S_X + kbB];
            #pragma unroll
            for (int i = 0; i < 4; ++i) { b.u[i] = pb[i]; b.u[i + 4] = pb[i + 8]; }
            v8f z0 = {}, z1 = {};
            h0acc[j] = __builtin_amdgcn_wmma_f32_16x16x32_f16(
                false, xa0.v, false, b.v, (short)0, z0, false, false);
            h1acc[j] = __builtin_amdgcn_wmma_f32_16x16x32_f16(
                false, xa1.v, false, b.v, (short)0, z1, false, false);
        }
        #pragma unroll
        for (int j = 0; j < 8; ++j) {
            int col = j * 16 + bcol;
            float bias = sB1[col];
            #pragma unroll
            for (int r = 0; r < 8; ++r) {
                float v0 = h0acc[j][r] + bias; v0 = v0 > 0.f ? v0 : 0.f;
                float v1 = h1acc[j][r] + bias; v1 = v1 > 0.f ? v1 : 0.f;
                sA[(crow0 + r)       * S_BT + col] = (_Float16)v0;
                sA[(crow0 + r + 128) * S_BT + col] = (_Float16)v1;
            }
        }
    }
    asm volatile("" ::: "memory");  // keep LDS store->load program order (HW is in-order per wave)

    // ================= Phase 2: layer 2 (WMMA, K=128) =================
    {
        v8f acc0[8] = {};
        v8f acc1[8] = {};
        #pragma unroll
        for (int kk = 0; kk < H_DIM; kk += 32) {
            Frag16 a0, a1;
            const unsigned int* pa0 = (const unsigned int*)&sA[mrow * S_BT + kk + klo];
            const unsigned int* pa1 = (const unsigned int*)&sA[(mrow + 128) * S_BT + kk + klo];
            #pragma unroll
            for (int i = 0; i < 4; ++i) {
                a0.u[i] = pa0[i]; a0.u[i + 4] = pa0[i + 8];
                a1.u[i] = pa1[i]; a1.u[i + 4] = pa1[i + 8];
            }
            #pragma unroll
            for (int j = 0; j < 8; ++j) {
                Frag16 b;
                const unsigned int* pb =
                    (const unsigned int*)&sBt[(j * 16 + bcol) * S_BT + kk + kbB];
                #pragma unroll
                for (int i = 0; i < 4; ++i) { b.u[i] = pb[i]; b.u[i + 4] = pb[i + 8]; }
                acc0[j] = __builtin_amdgcn_wmma_f32_16x16x32_f16(
                    false, a0.v, false, b.v, (short)0, acc0[j], false, false);
                acc1[j] = __builtin_amdgcn_wmma_f32_16x16x32_f16(
                    false, a1.v, false, b.v, (short)0, acc1[j], false, false);
            }
        }
        // bias + relu; overwrite own bands of sA with h2 (all reads done above)
        #pragma unroll
        for (int j = 0; j < 8; ++j) {
            int col = j * 16 + bcol;
            float bias = sB2[col];
            #pragma unroll
            for (int r = 0; r < 8; ++r) {
                float v0 = acc0[j][r] + bias; v0 = v0 > 0.f ? v0 : 0.f;
                float v1 = acc1[j][r] + bias; v1 = v1 > 0.f ? v1 : 0.f;
                sA[(crow0 + r)       * S_BT + col] = (_Float16)v0;
                sA[(crow0 + r + 128) * S_BT + col] = (_Float16)v1;
            }
        }
    }
    asm volatile("" ::: "memory");

    // ================= Phase 3: layer 3 (WMMA, N padded to 16) ========
    {
        v8f o0 = {}, o1 = {};
        #pragma unroll
        for (int kk = 0; kk < H_DIM; kk += 32) {
            Frag16 a0, a1, b;
            const unsigned int* pa0 = (const unsigned int*)&sA[mrow * S_BT + kk + klo];
            const unsigned int* pa1 = (const unsigned int*)&sA[(mrow + 128) * S_BT + kk + klo];
            const unsigned int* pb  = (const unsigned int*)&sW3t[bcol * S_BT + kk + kbB];
            #pragma unroll
            for (int i = 0; i < 4; ++i) {
                a0.u[i] = pa0[i]; a0.u[i + 4] = pa0[i + 8];
                a1.u[i] = pa1[i]; a1.u[i + 4] = pa1[i + 8];
            }
            #pragma unroll
            for (int i = 0; i < 4; ++i) { b.u[i] = pb[i]; b.u[i + 4] = pb[i + 8]; }
            o0 = __builtin_amdgcn_wmma_f32_16x16x32_f16(
                false, a0.v, false, b.v, (short)0, o0, false, false);
            o1 = __builtin_amdgcn_wmma_f32_16x16x32_f16(
                false, a1.v, false, b.v, (short)0, o1, false, false);
        }
        // D layout: column (lane&15) == output channel o; predicated store
        const int o = bcol;
        if (o < DOUT) {
            float bias = sB3[o];
            const long long base = (long long)p * N + row0 + crow0;
            #pragma unroll
            for (int r = 0; r < 8; ++r) {
                out[(base + r)       * DOUT + o] = o0[r] + bias;
                out[(base + r + 128) * DOUT + o] = o1[r] + bias;
            }
        }
    }
}

template<int DIN, int DOUT>
static void launch_attr(void* const* din, float* out, const _Float16* w2h,
                        int N, hipStream_t stream)
{
    dim3 grid(N / M_TILE, P_EXPERTS);
    if (w2h) {
        moe_mlp_fused<DIN, DOUT, true><<<grid, dim3(NTHREADS), 0, stream>>>(
            (const float*)din[0], (const float*)din[1], (const float*)din[2],
            (const float*)din[3], (const float*)din[4], (const float*)din[5],
            (const float*)din[6], w2h, out, N);
    } else {
        moe_mlp_fused<DIN, DOUT, false><<<grid, dim3(NTHREADS), 0, stream>>>(
            (const float*)din[0], (const float*)din[1], (const float*)din[2],
            (const float*)din[3], (const float*)din[4], (const float*)din[5],
            (const float*)din[6], (const _Float16*)nullptr, out, N);
    }
}

extern "C" void kernel_launch(void* const* d_in, const int* in_sizes, int n_in,
                              void* d_out, int out_size, void* d_ws, size_t ws_size,
                              hipStream_t stream)
{
    (void)n_in; (void)out_size;
    const int N = in_sizes[0] / 3;   // means is [N, 3]
    float* out = (float*)d_out;

    const int W2_ELEMS = P_EXPERTS * H_DIM * H_DIM;            // per attribute
    const size_t ws_need = (size_t)5 * W2_ELEMS * sizeof(_Float16);
    _Float16* w2h = (ws_size >= ws_need) ? (_Float16*)d_ws : nullptr;

    if (w2h) {
        dim3 cb(256), cg((W2_ELEMS + 255) / 256);
        for (int a = 0; a < 5; ++a)
            cvt_w2_f16<<<cg, cb, 0, stream>>>((const float*)d_in[a * 7 + 3],
                                              w2h + (size_t)a * W2_ELEMS, W2_ELEMS);
    }

    long long off = 0;
    // Inputs per attribute a: d_in[a*7 + {0:x, 1:W1, 2:b1, 3:W2, 4:b2, 5:W3, 6:b3}]
    launch_attr<3, 3>(d_in + 0 * 7, out + off, w2h ? w2h + 0LL * W2_ELEMS : nullptr, N, stream);
    off += (long long)P_EXPERTS * N * 3;
    launch_attr<3, 3>(d_in + 1 * 7, out + off, w2h ? w2h + 1LL * W2_ELEMS : nullptr, N, stream);
    off += (long long)P_EXPERTS * N * 3;
    launch_attr<3, 3>(d_in + 2 * 7, out + off, w2h ? w2h + 2LL * W2_ELEMS : nullptr, N, stream);
    off += (long long)P_EXPERTS * N * 3;
    launch_attr<4, 4>(d_in + 3 * 7, out + off, w2h ? w2h + 3LL * W2_ELEMS : nullptr, N, stream);
    off += (long long)P_EXPERTS * N * 4;
    launch_attr<1, 1>(d_in + 4 * 7, out + off, w2h ? w2h + 4LL * W2_ELEMS : nullptr, N, stream);
}
